// GATEncoder_61830349193582
// MI455X (gfx1250) — matrix operand order repaired and checked by reading, not compiled, not run on hardware
//
#include <hip/hip_runtime.h>
#include <hip/hip_bf16.h>

// ---------------------------------------------------------------------------
// GAT encoder for MI455X (gfx1250, wave32).
// Dense stages use v_wmma_f32_16x16x32_f16 (f16 operands, f32 accumulate).
// Edge phase uses L2-resident atomics (all node-feature arrays ~26MB << 192MB L2).
// ---------------------------------------------------------------------------

typedef __attribute__((ext_vector_type(16))) _Float16 v16h;
typedef __attribute__((ext_vector_type(8)))  _Float16 v8h;
typedef __attribute__((ext_vector_type(8)))  float    v8f;

#define NEG_SLOPE 0.2f
#define BN_EPS 1e-5f
#define NEG_INF_MONO 0x007FFFFFu   // monotone encoding of -inf

// order-preserving float<->uint mapping (u1 > u2  <=>  f1 > f2)
__device__ __forceinline__ unsigned fmono(float f) {
    unsigned b = __float_as_uint(f);
    return (b & 0x80000000u) ? ~b : (b | 0x80000000u);
}
__device__ __forceinline__ float fdemono(unsigned u) {
    return __uint_as_float((u & 0x80000000u) ? (u & 0x7FFFFFFFu) : ~u);
}
__device__ __forceinline__ float elu1(float x)   { return x > 0.f ? x : expm1f(x); }
__device__ __forceinline__ float lrelu(float x)  { return x > 0.f ? x : NEG_SLOPE * x; }

// ---------------------------------------------------------------------------
// Generic fill
// ---------------------------------------------------------------------------
__global__ void k_fill(unsigned* p, unsigned v, int n) {
    int t = blockIdx.x * blockDim.x + threadIdx.x;
    if (t < n) p[t] = v;
}

// ---------------------------------------------------------------------------
// Swizzle a row-major [K, Ncols] f32 weight into the WMMA B-operand layout:
// out[((nt*(K/32)+kb)*32 + lane)*16 + e] = W[k*Ncols + col]
//   col = nt*16 + (lane&15),  k = kb*32 + (e>>3)*16 + (lane>>4)*8 + (e&7)
// ---------------------------------------------------------------------------
__global__ void k_prepB(const float* __restrict__ W, _Float16* __restrict__ out,
                        int K, int Ncols) {
    int t = blockIdx.x * blockDim.x + threadIdx.x;
    if (t >= K * Ncols) return;
    int e    = t & 15;
    int lane = (t >> 4) & 31;
    int blk  = t >> 9;
    int kblocks = K >> 5;
    int kb = blk % kblocks;
    int nt = blk / kblocks;
    int col = nt * 16 + (lane & 15);
    int k   = kb * 32 + (e >> 3) * 16 + (lane >> 4) * 8 + (e & 7);
    out[t] = (_Float16)W[k * Ncols + col];
}

// ---------------------------------------------------------------------------
// WMMA GEMM: C[M x NCOLS] = A[M x K] (f16, row-major) * Bsw (+ bias)
// One wave computes a 16-row stripe. M padded to multiple of 16.
// ---------------------------------------------------------------------------
template <int K, int NCOLS>
__global__ void gemm_wmma(const _Float16* __restrict__ A,
                          const _Float16* __restrict__ Bsw,
                          const float* __restrict__ bias,
                          float* __restrict__ C, int Mtiles) {
    int wave = (blockIdx.x * blockDim.x + threadIdx.x) >> 5;
    int lane = threadIdx.x & 31;
    if (wave >= Mtiles) return;
    const int rowBase = wave * 16;
    const int m16 = lane & 15;
    const int khi = lane >> 4;
    const _Float16* Arow = A + (size_t)(rowBase + m16) * K + khi * 8;
#pragma unroll
    for (int nt = 0; nt < NCOLS / 16; ++nt) {
        v8f acc = {};
#pragma unroll
        for (int kb = 0; kb < K / 32; ++kb) {
            union { v16h v; v8h h[2]; } a;
            a.h[0] = *(const v8h*)(Arow + kb * 32);        // K = kb*32 + khi*8 + [0..7]
            a.h[1] = *(const v8h*)(Arow + kb * 32 + 16);   // K = +16
            v16h b = *(const v16h*)(Bsw + ((size_t)(nt * (K / 32) + kb) * 32 + lane) * 16);
            acc = __builtin_amdgcn_wmma_f32_16x16x32_f16(
                false, a.v, false, b, (short)0, acc, false, false);
        }
        int col = nt * 16 + m16;
        float bv = bias ? bias[col] : 0.0f;
#pragma unroll
        for (int r = 0; r < 8; ++r) {
            int m = rowBase + r + khi * 8;
            C[(size_t)m * NCOLS + col] = acc[r] + bv;
        }
    }
}

// ---------------------------------------------------------------------------
// h = elu(x @ W_emb + b_emb);  writes f32 + f16 copies; zeros pad rows.
// ---------------------------------------------------------------------------
__global__ void k_embed(const float* __restrict__ x, const float* __restrict__ W,
                        const float* __restrict__ b, int N, int Npad,
                        float* __restrict__ h, _Float16* __restrict__ hA) {
    int t = blockIdx.x * blockDim.x + threadIdx.x;
    if (t >= Npad * 64) return;
    int n = t >> 6, c = t & 63;
    float y = 0.f;
    if (n < N) {
        y = b[c];
#pragma unroll
        for (int i = 0; i < 4; ++i) y += x[n * 4 + i] * W[i * 64 + c];
        y = elu1(y);
    }
    h[t] = y;
    hA[t] = (_Float16)y;
}

// ---------------------------------------------------------------------------
// Attention scalars: al_src[n,h] = sum_c xh[n,h,c]*a_src[h,c] (same for dst)
// ---------------------------------------------------------------------------
__global__ void k_alpha(const float* __restrict__ xh, const float* __restrict__ aS,
                        const float* __restrict__ aD, int N, int H, int C,
                        float* __restrict__ alS, float* __restrict__ alD) {
    int t = blockIdx.x * blockDim.x + threadIdx.x;
    if (t >= N * H) return;
    int n = t / H, hd = t - n * H;
    const float* xr = xh + (size_t)n * 64 + hd * C;
    float s1 = 0.f, s2 = 0.f;
    for (int c = 0; c < C; ++c) {
        float v = xr[c];
        s1 += v * aS[hd * C + c];
        s2 += v * aD[hd * C + c];
    }
    alS[t] = s1;
    alD[t] = s2;
}

__device__ __forceinline__ void edge_sd(const int* ei, int E, int e, int& s, int& d) {
    if (e < E) { s = ei[e]; d = ei[E + e]; } else { s = d = e - E; }
}

// segment max of leaky_relu(al_src[s]+al_dst[d]) over dst
__global__ void k_edge_max(const int* __restrict__ ei, int E, int N, int H,
                           const float* __restrict__ alS, const float* __restrict__ alD,
                           unsigned* __restrict__ mx) {
    int e = blockIdx.x * blockDim.x + threadIdx.x;
    if (e >= E + N) return;
    int s, d; edge_sd(ei, E, e, s, d);
    for (int hd = 0; hd < H; ++hd) {
        float v = lrelu(alS[s * H + hd] + alD[d * H + hd]);
        atomicMax(&mx[d * H + hd], fmono(v));
    }
}

// segment sum of exp(e - max)
__global__ void k_edge_expsum(const int* __restrict__ ei, int E, int N, int H,
                              const float* __restrict__ alS, const float* __restrict__ alD,
                              const unsigned* __restrict__ mx, float* __restrict__ denom) {
    int e = blockIdx.x * blockDim.x + threadIdx.x;
    if (e >= E + N) return;
    int s, d; edge_sd(ei, E, e, s, d);
    for (int hd = 0; hd < H; ++hd) {
        float v = lrelu(alS[s * H + hd] + alD[d * H + hd]);
        float ex = expf(v - fdemono(mx[d * H + hd]));
        atomicAdd(&denom[d * H + hd], ex);
    }
}

// scatter-add: agg[dst] += alpha * xh[src]; 16 threads/edge, 4 channels each
__global__ void k_edge_agg(const int* __restrict__ ei, int E, int N, int H, int logC,
                           const float* __restrict__ alS, const float* __restrict__ alD,
                           const unsigned* __restrict__ mx, const float* __restrict__ denom,
                           const float* __restrict__ xh, float* __restrict__ agg) {
    int t = blockIdx.x * blockDim.x + threadIdx.x;
    int e = t >> 4;
    if (e >= E + N) return;
    int q = t & 15, c0 = q * 4;
    int s, d; edge_sd(ei, E, e, s, d);
    int hd = c0 >> logC;
    float v = lrelu(alS[s * H + hd] + alD[d * H + hd]);
    float alpha = expf(v - fdemono(mx[d * H + hd])) / (denom[d * H + hd] + 1e-16f);
    const float4 xv = *(const float4*)(xh + (size_t)s * 64 + c0);
    float* ag = agg + (size_t)d * 64 + c0;
    atomicAdd(ag + 0, alpha * xv.x);
    atomicAdd(ag + 1, alpha * xv.y);
    atomicAdd(ag + 2, alpha * xv.z);
    atomicAdd(ag + 3, alpha * xv.w);
}

// per-channel sum / sumsq of (agg + bias) for BN (train-mode, biased var)
__global__ void k_bnstats(const float* __restrict__ agg, const float* __restrict__ b,
                          int N, float* __restrict__ stats) {
    int tid = blockIdx.x * blockDim.x + threadIdx.x;
    int c = tid & 63;
    int rowstride = (gridDim.x * blockDim.x) >> 6;
    float s = 0.f, ss = 0.f;
    for (int n = tid >> 6; n < N; n += rowstride) {
        float v = agg[(size_t)n * 64 + c] + b[c];
        s += v; ss += v * v;
    }
    atomicAdd(&stats[c], s);
    atomicAdd(&stats[64 + c], ss);
}

// BN(train) + ELU; writes f32 + f16 copies (pad rows zeroed)
__global__ void k_bn_elu(const float* __restrict__ agg, const float* __restrict__ b,
                         const float* __restrict__ g, const float* __restrict__ be,
                         const float* __restrict__ stats, int N, int Npad,
                         float* __restrict__ h, _Float16* __restrict__ hA) {
    int t = blockIdx.x * blockDim.x + threadIdx.x;
    if (t >= Npad * 64) return;
    int n = t >> 6, c = t & 63;
    if (n >= N) { h[t] = 0.f; hA[t] = (_Float16)0.f; return; }
    float invN = 1.0f / (float)N;
    float mu  = stats[c] * invN;
    float var = stats[64 + c] * invN - mu * mu;
    float v = agg[t] + b[c];
    float y = g[c] * (v - mu) * rsqrtf(var + BN_EPS) + be[c];
    y = elu1(y);
    h[t] = y;
    hA[t] = (_Float16)y;
}

// global max/mean pool by graph id (atomics into [256 x 64])
__global__ void k_pool(const float* __restrict__ h, const int* __restrict__ batch, int N,
                       unsigned* __restrict__ pmax, float* __restrict__ psum,
                       float* __restrict__ counts) {
    int t = blockIdx.x * blockDim.x + threadIdx.x;
    if (t >= N * 64) return;
    int n = t >> 6, c = t & 63;
    int g = batch[n];
    float v = h[t];
    atomicMax(&pmax[g * 64 + c], fmono(v));
    atomicAdd(&psum[g * 64 + c], v);
    if (c == 0) atomicAdd(&counts[g], 1.0f);
}

// comb = [max | mean] per graph, f32 + f16
__global__ void k_comb(const unsigned* __restrict__ pmax, const float* __restrict__ psum,
                       const float* __restrict__ counts, float* __restrict__ comb,
                       _Float16* __restrict__ combH) {
    int t = blockIdx.x * blockDim.x + threadIdx.x;
    if (t >= 256 * 128) return;
    int g = t >> 7, c = t & 127;
    float cnt = counts[g];
    float v;
    if (c < 64) v = (cnt > 0.f) ? fdemono(pmax[g * 64 + c]) : 0.f;
    else        v = psum[g * 64 + (c - 64)] / fmaxf(cnt, 1.f);
    comb[t] = v;
    combH[t] = (_Float16)v;
}

// ---------------------------------------------------------------------------
// Host launcher
// ---------------------------------------------------------------------------
static inline int cdiv(long long a, int b) { return (int)((a + b - 1) / b); }

extern "C" void kernel_launch(void* const* d_in, const int* in_sizes, int n_in,
                              void* d_out, int out_size, void* d_ws, size_t ws_size,
                              hipStream_t stream) {
    const float* x      = (const float*)d_in[0];
    const int*   ei     = (const int*)  d_in[1];
    const int*   batch  = (const int*)  d_in[2];
    const float* W_emb  = (const float*)d_in[3];
    const float* b_emb  = (const float*)d_in[4];
    const float* W1     = (const float*)d_in[5];
    const float* a1_src = (const float*)d_in[6];
    const float* a1_dst = (const float*)d_in[7];
    const float* b1     = (const float*)d_in[8];
    const float* g1     = (const float*)d_in[9];
    const float* be1    = (const float*)d_in[10];
    const float* W2     = (const float*)d_in[11];
    const float* a2_src = (const float*)d_in[12];
    const float* a2_dst = (const float*)d_in[13];
    const float* b2     = (const float*)d_in[14];
    const float* g2     = (const float*)d_in[15];
    const float* be2    = (const float*)d_in[16];
    const float* W_out  = (const float*)d_in[17];
    const float* b_out  = (const float*)d_in[18];
    float* out = (float*)d_out;

    const int N = in_sizes[2];
    const int E = in_sizes[1] / 2;
    const int Npad = (N + 15) & ~15;
    const int Etot = E + N;

    // --- workspace bump allocator (256B aligned) ---
    char* ws = (char*)d_ws;
    size_t off = 0;
    auto alloc = [&](size_t bytes) -> void* {
        void* p = ws + off;
        off = (off + bytes + 255) & ~(size_t)255;
        return p;
    };
    _Float16* hA    = (_Float16*)alloc((size_t)Npad * 64 * 2);
    float*    h     = (float*)   alloc((size_t)Npad * 64 * 4);
    float*    xh    = (float*)   alloc((size_t)Npad * 64 * 4);
    float*    agg   = (float*)   alloc((size_t)Npad * 64 * 4);
    float*    alS   = (float*)   alloc((size_t)N * 2 * 4);
    float*    alD   = (float*)   alloc((size_t)N * 2 * 4);
    unsigned* mx    = (unsigned*)alloc((size_t)N * 2 * 4);
    float*    denom = (float*)   alloc((size_t)N * 2 * 4);
    float*    stats = (float*)   alloc(128 * 4);
    _Float16* Bsw1  = (_Float16*)alloc(64 * 64 * 2);
    _Float16* Bsw2  = (_Float16*)alloc(64 * 64 * 2);
    _Float16* BswO  = (_Float16*)alloc(128 * 128 * 2);
    unsigned* pmax  = (unsigned*)alloc(256 * 64 * 4);
    float*    psum  = (float*)   alloc(256 * 64 * 4);
    float*    cnts  = (float*)   alloc(256 * 4);
    float*    comb  = (float*)   alloc(256 * 128 * 4);
    _Float16* combH = (_Float16*)alloc(256 * 128 * 2);
    (void)ws_size; (void)n_in; (void)out_size;

    const int B = 256;

    // weight swizzles for WMMA B operand
    k_prepB<<<cdiv(64 * 64, B), B, 0, stream>>>(W1, Bsw1, 64, 64);
    k_prepB<<<cdiv(64 * 64, B), B, 0, stream>>>(W2, Bsw2, 64, 64);
    k_prepB<<<cdiv(128 * 128, B), B, 0, stream>>>(W_out, BswO, 128, 128);

    // embedding + ELU
    k_embed<<<cdiv((long long)Npad * 64, B), B, 0, stream>>>(x, W_emb, b_emb, N, Npad, h, hA);

    const int Mtiles = Npad / 16;
    const int gemmBlocks = cdiv((long long)Mtiles * 32, B);

    auto conv = [&](const _Float16* Bsw, const float* aS, const float* aD,
                    const float* bias, const float* gam, const float* bet,
                    int H, int C, int logC) {
        k_fill<<<cdiv(N * H, B), B, 0, stream>>>(mx, NEG_INF_MONO, N * H);
        k_fill<<<cdiv(N * H, B), B, 0, stream>>>((unsigned*)denom, 0u, N * H);
        k_fill<<<cdiv((long long)N * 64, B), B, 0, stream>>>((unsigned*)agg, 0u, N * 64);
        k_fill<<<1, 128, 0, stream>>>((unsigned*)stats, 0u, 128);

        gemm_wmma<64, 64><<<gemmBlocks, B, 0, stream>>>(hA, Bsw, nullptr, xh, Mtiles);

        k_alpha<<<cdiv(N * H, B), B, 0, stream>>>(xh, aS, aD, N, H, C, alS, alD);
        k_edge_max<<<cdiv(Etot, B), B, 0, stream>>>(ei, E, N, H, alS, alD, mx);
        k_edge_expsum<<<cdiv(Etot, B), B, 0, stream>>>(ei, E, N, H, alS, alD, mx, denom);
        k_edge_agg<<<cdiv((long long)Etot * 16, B), B, 0, stream>>>(
            ei, E, N, H, logC, alS, alD, mx, denom, xh, agg);

        k_bnstats<<<512, B, 0, stream>>>(agg, bias, N, stats);
        k_bn_elu<<<cdiv((long long)Npad * 64, B), B, 0, stream>>>(
            agg, bias, gam, bet, stats, N, Npad, h, hA);
    };

    // conv1: heads=2, out_ch=32 ; conv2: heads=1, out_ch=64
    conv(Bsw1, a1_src, a1_dst, b1, g1, be1, 2, 32, 5);
    conv(Bsw2, a2_src, a2_dst, b2, g2, be2, 1, 64, 6);

    // global max/mean pool
    k_fill<<<cdiv(256 * 64, B), B, 0, stream>>>(pmax, NEG_INF_MONO, 256 * 64);
    k_fill<<<cdiv(256 * 64, B), B, 0, stream>>>((unsigned*)psum, 0u, 256 * 64);
    k_fill<<<1, 256, 0, stream>>>((unsigned*)cnts, 0u, 256);
    k_pool<<<cdiv((long long)N * 64, B), B, 0, stream>>>(h, batch, N, pmax, psum, cnts);
    k_comb<<<cdiv(256 * 128, B), B, 0, stream>>>(pmax, psum, cnts, comb, combH);

    // output projection: [256,128] @ [128,128] + b_out  (16 waves = 1 block)
    gemm_wmma<128, 128><<<1, 512, 0, stream>>>(combH, BswO, b_out, out, 16);
}
